// SelfAttention_74689481277598
// MI455X (gfx1250) — matrix operand order repaired and checked by reading, not compiled, not run on hardware
//
#include <hip/hip_runtime.h>
#include <hip/hip_bf16.h>

// Problem constants (fixed by the reference)
#define BATCH   4
#define S_LEN   2048
#define NHEAD   16
#define HDIM    64
#define DMODEL  1024
#define MROWS   (BATCH * S_LEN)   // 8192

typedef __attribute__((ext_vector_type(16))) __bf16 v16bf;
typedef __attribute__((ext_vector_type(8)))  float  v8f;

__device__ __forceinline__ v16bf make_frag(uint4 lo, uint4 hi) {
  union { v16bf v; uint4 u[2]; } r;
  r.u[0] = lo; r.u[1] = hi;
  return r.v;
}

// ---------------------------------------------------------------------------
// f32 -> bf16 elementwise convert (vectorized: 4 floats -> 8 bytes per thread)
// ---------------------------------------------------------------------------
__global__ __launch_bounds__(256) void f32_to_bf16_vec(const float* __restrict__ in,
                                                       __bf16* __restrict__ out,
                                                       int n4) {
  int i = blockIdx.x * blockDim.x + threadIdx.x;
  if (i >= n4) return;
  float4 f = ((const float4*)in)[i];
  union { __bf16 b[4]; uint2 u; } r;
  r.b[0] = (__bf16)f.x; r.b[1] = (__bf16)f.y;
  r.b[2] = (__bf16)f.z; r.b[3] = (__bf16)f.w;
  ((uint2*)out)[i] = r.u;
}

// ---------------------------------------------------------------------------
// Tiled WMMA GEMM:  Y[M,N] = X[M,K] * W[N,K]^T + bias[N]
// Block tile 128x256, 256 threads = 8 waves (2x4), wave tile 64x64
// (4x4 grid of 16x16 WMMA tiles -> 16 wmma per 8 fragment loads per K32).
// Double-buffered LDS staging, one barrier per K-step, global loads for the
// next tile issued before the current tile's math.
// MODE 0: f32 row-major [M,N]          (final output projection)
// MODE 1: bf16 [b, h, s, d]            (Q, K for attention)
// MODE 2: bf16 [b, h, d, s] transposed (V for attention: A*V B-operand layout)
// ---------------------------------------------------------------------------
template <int MODE>
__global__ __launch_bounds__(256, 1) void gemm_bf16_wmma(
    const __bf16* __restrict__ X, const __bf16* __restrict__ W,
    const float* __restrict__ bias, void* __restrict__ outp,
    int M, int N, int K) {
  constexpr int LSTR = 40;  // 32 + 8 pad (keeps 16B alignment, staggers banks)
  __shared__ __attribute__((aligned(16))) __bf16 As[2][128 * LSTR];
  __shared__ __attribute__((aligned(16))) __bf16 Bs[2][256 * LSTR];

  const int tid  = threadIdx.x;
  const int wave = tid >> 5, lane = tid & 31;
  const int half = lane >> 4, ln = lane & 15;
  const int wm = (wave >> 2) * 64;   // wave row offset in block tile (0/64)
  const int wn = (wave & 3) * 64;    // wave col offset in block tile (0..192)
  const int m0 = blockIdx.y * 128;
  const int n0 = blockIdx.x * 256;

  // Staging: 128x32 X-tile + 256x32 W-tile = 1536 16B chunks / 256 threads.
  uint4 stg[6];
  auto glb_load = [&](int k) {
    for (int it = 0; it < 6; ++it) {
      int idx = tid + it * 256;        // 0..1535
      int row = idx >> 2;              // 0..383
      int ch  = (idx & 3) * 8;
      const __bf16* p = (row < 128)
          ? X + (size_t)(m0 + row) * K + k + ch
          : W + (size_t)(n0 + row - 128) * K + k + ch;
      stg[it] = *(const uint4*)p;
    }
  };
  auto lds_store = [&](int buf) {
    for (int it = 0; it < 6; ++it) {
      int idx = tid + it * 256;
      int row = idx >> 2;
      int ch  = (idx & 3) * 8;
      if (row < 128) *(uint4*)(&As[buf][row * LSTR + ch]) = stg[it];
      else           *(uint4*)(&Bs[buf][(row - 128) * LSTR + ch]) = stg[it];
    }
  };

  glb_load(0);
  lds_store(0);
  __syncthreads();

  v8f acc[4][4] = {};

  for (int k0 = 0; k0 < K; k0 += 32) {
    const int cur = (k0 >> 5) & 1;
    const bool more = (k0 + 32) < K;
    if (more) glb_load(k0 + 32);   // overlap with the 16 WMMAs below

    // A fragments: lane holds row (ln), K elems {kh*8..+8, 16+kh*8..+8}
    v16bf af[4];
    for (int mt = 0; mt < 4; ++mt) {
      const __bf16* p = &As[cur][(wm + mt * 16 + ln) * LSTR + half * 8];
      af[mt] = make_frag(*(const uint4*)p, *(const uint4*)(p + 16));
    }
    // B fragments (K x N, element (k,n) = W[n][k]): lane holds col (ln),
    // K elems {kh*8..+8, 16+kh*8..+8}.
    for (int nt = 0; nt < 4; ++nt) {
      const __bf16* p = &Bs[cur][(wn + nt * 16 + ln) * LSTR + half * 8];
      v16bf bf = make_frag(*(const uint4*)p, *(const uint4*)(p + 16));
      for (int mt = 0; mt < 4; ++mt)
        acc[mt][nt] = __builtin_amdgcn_wmma_f32_16x16x32_bf16(
            false, af[mt], false, bf, (short)0, acc[mt][nt], false, false);
    }

    if (more) lds_store(cur ^ 1);
    __syncthreads();  // next iter reads buf cur^1; iter after writes buf cur
  }

  // Epilogue: C layout is VGPR i <-> row i + 8*half, col = ln.
  for (int mt = 0; mt < 4; ++mt)
    for (int nt = 0; nt < 4; ++nt) {
      int gn = n0 + wn + nt * 16 + ln;
      float bv = bias[gn];
      for (int i = 0; i < 8; ++i) {
        int gm = m0 + wm + mt * 16 + i + 8 * half;
        float y = acc[mt][nt][i] + bv;
        if (MODE == 0) {
          ((float*)outp)[(size_t)gm * N + gn] = y;
        } else {
          int b = gm >> 11, s = gm & (S_LEN - 1);  // S = 2048
          int h = gn >> 6,  d = gn & (HDIM - 1);   // HD = 64
          size_t off;
          if (MODE == 1)
            off = (((size_t)(b * NHEAD + h)) * S_LEN + s) * HDIM + d;
          else
            off = (((size_t)(b * NHEAD + h)) * HDIM + d) * S_LEN + s;
          ((__bf16*)outp)[off] = (__bf16)y;
        }
      }
    }
}

// ---------------------------------------------------------------------------
// Flash attention (causal, streaming softmax).
// Grid: B*H*(S/64) blocks; 128 threads = 4 waves; wave handles 16 q-rows.
// Q,K: bf16 [b,h,s,d].  V: bf16 [b,h,d,s] (pre-transposed).
// ctx out: bf16 [b, s, h*64+d]  (row-major operand for O-projection GEMM).
// ---------------------------------------------------------------------------
__global__ __launch_bounds__(128, 4) void attn_fmha(
    const __bf16* __restrict__ Q, const __bf16* __restrict__ Kd,
    const __bf16* __restrict__ Vt, __bf16* __restrict__ ctx) {
  __shared__ __attribute__((aligned(16))) __bf16 Plds[4][16 * 32];

  const int lane = threadIdx.x & 31, wave = threadIdx.x >> 5;
  const int half = lane >> 4, ln = lane & 15;
  const int qt = blockIdx.x & 31;        // S/64 = 32 q-tiles
  const int bh = blockIdx.x >> 5;        // 0 .. B*H-1
  const int q0 = qt * 64 + wave * 16;    // first q row of this wave

  const __bf16* Qb = Q  + (size_t)bh * S_LEN * HDIM;
  const __bf16* Kb = Kd + (size_t)bh * S_LEN * HDIM;
  const __bf16* Vb = Vt + (size_t)bh * S_LEN * HDIM;  // [HDIM][S_LEN]

  // Load Q fragments once (HD=64 -> two K32 steps).
  v16bf qf[2];
  {
    const __bf16* qrow = Qb + (size_t)(q0 + ln) * HDIM;
    for (int kk = 0; kk < 2; ++kk) {
      const __bf16* p = qrow + kk * 32 + half * 8;
      qf[kk] = make_frag(*(const uint4*)p, *(const uint4*)(p + 16));
    }
  }

  v8f acc[4] = {};           // output: 4 N-tiles of 16 covering HD=64
  float mrow[8], lrow[8];    // per-row running max / sum (row = i + 8*half)
  for (int i = 0; i < 8; ++i) { mrow[i] = -3e38f; lrow[i] = 0.f; }

  const int nkb = (q0 + 15) / 32 + 1;  // causal: only k-blocks <= diag
  for (int kb = 0; kb < nkb; ++kb) {
    // ---- scores: S = Q * K^T, two 16x16 tiles covering 32 k-cols ----
    v8f sacc[2] = {};
    for (int kt = 0; kt < 2; ++kt) {
      const __bf16* krow = Kb + (size_t)(kb * 32 + kt * 16 + ln) * HDIM;
      for (int kk = 0; kk < 2; ++kk) {
        const __bf16* p = krow + kk * 32 + half * 8;
        v16bf kf = make_frag(*(const uint4*)p, *(const uint4*)(p + 16));
        sacc[kt] = __builtin_amdgcn_wmma_f32_16x16x32_bf16(
            false, qf[kk], false, kf, (short)0, sacc[kt], false, false);
      }
    }
    // ---- causal mask + scale + online softmax update ----
    float p0[8], p1[8], corr[8];
    for (int i = 0; i < 8; ++i) {
      const int qr = q0 + i + 8 * half;
      const int c0 = kb * 32 + ln, c1 = c0 + 16;
      float s0 = (c0 <= qr) ? sacc[0][i] * 0.125f : -3e38f;  // 1/sqrt(64)
      float s1 = (c1 <= qr) ? sacc[1][i] * 0.125f : -3e38f;
      float mt = fmaxf(s0, s1);
      mt = fmaxf(mt, __shfl_xor(mt, 1));
      mt = fmaxf(mt, __shfl_xor(mt, 2));
      mt = fmaxf(mt, __shfl_xor(mt, 4));
      mt = fmaxf(mt, __shfl_xor(mt, 8));
      float mn = fmaxf(mrow[i], mt);
      float c  = __expf(mrow[i] - mn);
      float e0 = __expf(s0 - mn);
      float e1 = __expf(s1 - mn);
      float rs = e0 + e1;
      rs += __shfl_xor(rs, 1);
      rs += __shfl_xor(rs, 2);
      rs += __shfl_xor(rs, 4);
      rs += __shfl_xor(rs, 8);
      lrow[i] = lrow[i] * c + rs;
      mrow[i] = mn;
      corr[i] = c; p0[i] = e0; p1[i] = e1;
    }
    for (int nt = 0; nt < 4; ++nt)
      for (int i = 0; i < 8; ++i) acc[nt][i] *= corr[i];

    // ---- P (C-layout) -> LDS -> A-fragment layout ----
    for (int i = 0; i < 8; ++i) {
      int r = i + 8 * half;
      Plds[wave][r * 32 + ln]      = (__bf16)p0[i];
      Plds[wave][r * 32 + 16 + ln] = (__bf16)p1[i];
    }
    asm volatile("s_wait_dscnt 0x0" ::: "memory");  // per-wave LDS RAW
    v16bf pa;
    {
      const __bf16* p = &Plds[wave][ln * 32 + half * 8];
      pa = make_frag(*(const uint4*)p, *(const uint4*)(p + 16));
    }
    // ---- acc += P * V  (V pre-transposed: element (k=s, n=d) contiguous) ----
    for (int nt = 0; nt < 4; ++nt) {
      const __bf16* vrow = Vb + (size_t)(nt * 16 + ln) * S_LEN + kb * 32 + half * 8;
      v16bf vf = make_frag(*(const uint4*)vrow, *(const uint4*)(vrow + 16));
      acc[nt] = __builtin_amdgcn_wmma_f32_16x16x32_bf16(
          false, pa, false, vf, (short)0, acc[nt], false, false);
    }
  }

  // ---- normalize and store ctx[b, s, h*64+d] as bf16 ----
  float inv[8];
  for (int i = 0; i < 8; ++i) inv[i] = 1.0f / lrow[i];
  const int b = bh >> 4, h = bh & 15;
  for (int nt = 0; nt < 4; ++nt) {
    int d = nt * 16 + ln;
    for (int i = 0; i < 8; ++i) {
      int qr = q0 + i + 8 * half;
      ctx[((size_t)b * S_LEN + qr) * DMODEL + h * HDIM + d] =
          (__bf16)(acc[nt][i] * inv[i]);
    }
  }
}

// ---------------------------------------------------------------------------
// Launch
// ---------------------------------------------------------------------------
extern "C" void kernel_launch(void* const* d_in, const int* in_sizes, int n_in,
                              void* d_out, int out_size, void* d_ws, size_t ws_size,
                              hipStream_t stream) {
  (void)in_sizes; (void)n_in; (void)out_size; (void)ws_size;
  const float* x    = (const float*)d_in[0];
  const float* wq_w = (const float*)d_in[1];
  const float* wq_b = (const float*)d_in[2];
  const float* wk_w = (const float*)d_in[3];
  const float* wk_b = (const float*)d_in[4];
  const float* wv_w = (const float*)d_in[5];
  const float* wv_b = (const float*)d_in[6];
  const float* wo_w = (const float*)d_in[7];
  const float* wo_b = (const float*)d_in[8];

  char* ws = (char*)d_ws;
  const size_t MB = 1024 * 1024;
  __bf16* xb  = (__bf16*)(ws);            // 16 MB: x as bf16 [8192,1024]
  __bf16* wqb = (__bf16*)(ws + 16 * MB);  // 2 MB each
  __bf16* wkb = (__bf16*)(ws + 18 * MB);
  __bf16* wvb = (__bf16*)(ws + 20 * MB);
  __bf16* wob = (__bf16*)(ws + 22 * MB);
  __bf16* Qh  = (__bf16*)(ws + 24 * MB);  // 16 MB: [b,h,s,d]
  __bf16* Kh  = (__bf16*)(ws + 40 * MB);  // 16 MB: [b,h,s,d]
  __bf16* Vt  = (__bf16*)(ws + 56 * MB);  // 16 MB: [b,h,d,s]
  __bf16* ctx = (__bf16*)(ws + 72 * MB);  // 16 MB: [b,s,D]
  // total workspace: 88 MB (fits in the 192 MB L2)

  const int nx = BATCH * S_LEN * DMODEL;    // 8388608
  const int nw = DMODEL * DMODEL;           // 1048576
  f32_to_bf16_vec<<<(nx / 4) / 256, 256, 0, stream>>>(x,    xb,  nx / 4);
  f32_to_bf16_vec<<<(nw / 4) / 256, 256, 0, stream>>>(wq_w, wqb, nw / 4);
  f32_to_bf16_vec<<<(nw / 4) / 256, 256, 0, stream>>>(wk_w, wkb, nw / 4);
  f32_to_bf16_vec<<<(nw / 4) / 256, 256, 0, stream>>>(wv_w, wvb, nw / 4);
  f32_to_bf16_vec<<<(nw / 4) / 256, 256, 0, stream>>>(wo_w, wob, nw / 4);

  dim3 gg(DMODEL / 256, MROWS / 128);  // (4, 64)
  gemm_bf16_wmma<1><<<gg, 256, 0, stream>>>(xb, wqb, wq_b, Qh, MROWS, DMODEL, DMODEL);
  gemm_bf16_wmma<1><<<gg, 256, 0, stream>>>(xb, wkb, wk_b, Kh, MROWS, DMODEL, DMODEL);
  gemm_bf16_wmma<2><<<gg, 256, 0, stream>>>(xb, wvb, wv_b, Vt, MROWS, DMODEL, DMODEL);

  attn_fmha<<<BATCH * NHEAD * (S_LEN / 64), 128, 0, stream>>>(Qh, Kh, Vt, ctx);

  gemm_bf16_wmma<0><<<gg, 256, 0, stream>>>(ctx, wob, wo_b, d_out, MROWS, DMODEL, DMODEL);
}
